// Module_2765958939237
// MI455X (gfx1250) — compile-verified, matrix operand order checked by reading, not run on hardware
//
#include <hip/hip_runtime.h>
#include <hip/hip_bf16.h>
#include <math.h>

// ---------------------------------------------------------------------------
// Model constants (from the reference)
// ---------------------------------------------------------------------------
#define NNODE 50000
#define EMBD  128
#define BQ    64
#define SQ    200
#define BS    (BQ * SQ)      // 12800 rows
#define ST    13             // ceil(200/16) tiles
#define SROW  208            // padded score rows
#define SLD   224            // padded score row stride (7 * 32)

typedef __attribute__((ext_vector_type(16))) _Float16 v16h;
typedef __attribute__((ext_vector_type(8)))  float    v8f;

#define DEV static __device__ __forceinline__

DEV float wred_sum(float v) {
#pragma unroll
  for (int i = 16; i > 0; i >>= 1) v += __shfl_xor(v, i, 32);
  return v;
}
DEV float wred_max(float v) {
#pragma unroll
  for (int i = 16; i > 0; i >>= 1) v = fmaxf(v, __shfl_xor(v, i, 32));
  return v;
}

// ---------------------------------------------------------------------------
// WMMA fragment loaders: f32 memory -> f16 fragments, CDNA5 wave32 layouts.
// A (16x32, MxK): lanes 0-15 / 16-31 = K-halves; e<8 -> K=half*8+e,
//                 e>=8 -> K=16+half*8+(e-8).
// B (32x16, KxN): lane&15 = N column, same K striping as A across elements.
// ---------------------------------------------------------------------------
DEV v16h load_a_frag(const float* __restrict__ A, int lda, int row0, int kb,
                     int lane, int rowMax) {
  const int half = lane >> 4, m = lane & 15;
  const int row = row0 + m;
  v16h a;
  if (row < rowMax) {
    const float* p = A + (size_t)row * lda + kb + half * 8;
#pragma unroll
    for (int e = 0; e < 8; ++e) a[e] = (_Float16)p[e];
#pragma unroll
    for (int e = 0; e < 8; ++e) a[8 + e] = (_Float16)p[16 + e];
  } else {
#pragma unroll
    for (int e = 0; e < 16; ++e) a[e] = (_Float16)0.0f;
  }
  return a;
}

DEV v16h load_b_frag(const float* __restrict__ B, int ldb, int col0, int kb,
                     int lane, int kMax) {
  const int half = lane >> 4, n = lane & 15;
  const int col = col0 + n;
  v16h b;
#pragma unroll
  for (int e = 0; e < 16; ++e) {
    const int k = kb + ((e < 8) ? (half * 8 + e) : (16 + half * 8 + (e - 8)));
    const float v = (k < kMax) ? B[(size_t)k * ldb + col] : 0.0f;
    b[e] = (_Float16)v;
  }
  return b;
}

// Logical B[k,n] = Bt[n*ldb + k]  (i.e. B = Kmat^T loaded straight from rows)
DEV v16h load_bt_frag(const float* __restrict__ Bt, int ldb, int col0, int kb,
                      int lane, int colMax) {
  const int half = lane >> 4, n = lane & 15;
  const int col = col0 + n;
  v16h b;
  if (col < colMax) {
    const float* p = Bt + (size_t)col * ldb + kb + half * 8;
#pragma unroll
    for (int e = 0; e < 8; ++e) b[e] = (_Float16)p[e];
#pragma unroll
    for (int e = 0; e < 8; ++e) b[8 + e] = (_Float16)p[16 + e];
  } else {
#pragma unroll
    for (int e = 0; e < 16; ++e) b[e] = (_Float16)0.0f;
  }
  return b;
}

DEV v8f wmma16(v16h a, v16h b, v8f c) {
  return __builtin_amdgcn_wmma_f32_16x16x32_f16(false, a, false, b, (short)0, c,
                                                false, false);
}

// ---------------------------------------------------------------------------
// Generic GEMM: C[M,N] = act(A[M,K] @ B[K,N] + bias). M%16==N%16==K%32==0.
// One 16x16 tile per wave, 4 waves per block.
// ---------------------------------------------------------------------------
__global__ void k_gemm(const float* __restrict__ A, int lda,
                       const float* __restrict__ B, int ldb,
                       const float* __restrict__ bias,
                       float* __restrict__ C, int ldc,
                       int M, int N, int K, int act) {
  const int lane = threadIdx.x & 31;
  const int wave = blockIdx.x * (blockDim.x >> 5) + (threadIdx.x >> 5);
  const int tn = N >> 4;
  const int total = (M >> 4) * tn;
  if (wave >= total) return;                 // wave-uniform: EXEC stays full
  const int row0 = (wave / tn) << 4;
  const int col0 = (wave % tn) << 4;
  v8f acc = {};
  for (int kb = 0; kb < K; kb += 32) {
    v16h a = load_a_frag(A, lda, row0, kb, lane, 0x7fffffff);
    v16h b = load_b_frag(B, ldb, col0, kb, lane, 0x7fffffff);
    acc = wmma16(a, b, acc);
  }
  const int half = lane >> 4, n = lane & 15;
  const float bv = bias ? bias[col0 + n] : 0.0f;
#pragma unroll
  for (int r = 0; r < 8; ++r) {
    float v = acc[r] + bv;
    if (act == 1) v = tanhf(v);
    C[(size_t)(row0 + half * 8 + r) * ldc + col0 + n] = v;
  }
}

// ---------------------------------------------------------------------------
// Attention scores: Sc[b,h,q,k] = (Q @ K^T) / sqrt(dk). dk=32 -> one WMMA.
// ---------------------------------------------------------------------------
__global__ void k_scores(const float* __restrict__ Q, const float* __restrict__ Kx,
                         float* __restrict__ Sc, float scale) {
  const int lane = threadIdx.x & 31;
  const int wave = blockIdx.x * (blockDim.x >> 5) + (threadIdx.x >> 5);
  const int total = BQ * 4 * ST * ST;
  if (wave >= total) return;
  const int batch = wave / (ST * ST);
  const int t = wave % (ST * ST);
  const int b = batch >> 2, h = batch & 3;
  const int row0 = (t / ST) << 4, col0 = (t % ST) << 4;
  const float* Qb = Q + (size_t)b * SQ * EMBD + h * 32;
  const float* Kb = Kx + (size_t)b * SQ * EMBD + h * 32;
  v8f acc = {};
  v16h a = load_a_frag(Qb, EMBD, row0, 0, lane, SQ);
  v16h bt = load_bt_frag(Kb, EMBD, col0, 0, lane, SQ);
  acc = wmma16(a, bt, acc);
  float* Sb = Sc + (size_t)batch * SROW * SLD;
  const int half = lane >> 4, n = lane & 15;
#pragma unroll
  for (int r = 0; r < 8; ++r) {
    const int row = row0 + half * 8 + r;
    if (row < SQ) Sb[(size_t)row * SLD + col0 + n] = acc[r] * scale;
  }
}

// Row softmax over 200 keys (optional key-padding mask), zero-fills pad cols.
__global__ void k_softmax(float* __restrict__ Sc, const int* __restrict__ seq,
                          int useMask) {
  const int lane = threadIdx.x & 31;
  const int wave = blockIdx.x * (blockDim.x >> 5) + (threadIdx.x >> 5);
  const int total = BQ * 4 * SQ;
  if (wave >= total) return;
  const int batch = wave / SQ, row = wave % SQ;
  const int b = batch >> 2;
  float* p = Sc + (size_t)batch * SROW * SLD + (size_t)row * SLD;
  float vals[7];
  float mx = -3.0e38f;
#pragma unroll
  for (int i = 0; i < 7; ++i) {
    const int col = i * 32 + lane;
    float v = -3.0e38f;
    if (col < SQ) {
      v = p[col];
      if (useMask && seq[b * SQ + col] == 0) v += -1.0e9f;
      mx = fmaxf(mx, v);
    }
    vals[i] = v;
  }
  mx = wred_max(mx);
  float sum = 0.0f;
#pragma unroll
  for (int i = 0; i < 7; ++i) {
    const int col = i * 32 + lane;
    const float e = (col < SQ) ? expf(vals[i] - mx) : 0.0f;
    vals[i] = e;
    sum += e;
  }
  sum = wred_sum(sum);
  const float inv = 1.0f / sum;
#pragma unroll
  for (int i = 0; i < 7; ++i) p[i * 32 + lane] = vals[i] * inv;
}

// O[b,s,h*32+d] = P @ V. K padded to 224 (pad cols of P are zero).
__global__ void k_attnout(const float* __restrict__ Sc, const float* __restrict__ V,
                          float* __restrict__ O) {
  const int lane = threadIdx.x & 31;
  const int wave = blockIdx.x * (blockDim.x >> 5) + (threadIdx.x >> 5);
  const int total = BQ * 4 * ST * 2;
  if (wave >= total) return;
  const int batch = wave / (ST * 2);
  const int t = wave % (ST * 2);
  const int b = batch >> 2, h = batch & 3;
  const int row0 = (t >> 1) << 4, col0 = (t & 1) << 4;
  const float* Sb = Sc + (size_t)batch * SROW * SLD;
  const float* Vb = V + (size_t)b * SQ * EMBD + h * 32;
  v8f acc = {};
  for (int kb = 0; kb < SLD; kb += 32) {
    v16h a = load_a_frag(Sb, SLD, row0, kb, lane, SROW);
    v16h bf = load_b_frag(Vb, EMBD, col0, kb, lane, SQ);  // zero beyond S
    acc = wmma16(a, bf, acc);
  }
  const int half = lane >> 4, n = lane & 15;
#pragma unroll
  for (int r = 0; r < 8; ++r) {
    const int row = row0 + half * 8 + r;
    if (row < SQ)
      O[(size_t)(b * SQ + row) * EMBD + h * 32 + col0 + n] = acc[r];
  }
}

// ---------------------------------------------------------------------------
// SpMM segment-sum: out[row,:] += val * x[col,:].  One wave per edge,
// float4 per lane, atomic f32 adds (rows are random; x stays L2-resident).
// ---------------------------------------------------------------------------
__global__ void k_spmm(const int* __restrict__ row, const int* __restrict__ col,
                       const float* __restrict__ val, const float* __restrict__ x,
                       float* __restrict__ out, int nE) {
  const int g = blockIdx.x * blockDim.x + threadIdx.x;
  const int e = g >> 5;
  if (e >= nE) return;
  const int lane = g & 31;
  const int r = row[e], c = col[e];
  const float v = val[e];
  const float4 xv = ((const float4*)(x + (size_t)c * EMBD))[lane];
  float* op = out + (size_t)r * EMBD + lane * 4;
  atomicAdd(op + 0, v * xv.x);
  atomicAdd(op + 1, v * xv.y);
  atomicAdd(op + 2, v * xv.z);
  atomicAdd(op + 3, v * xv.w);
}

// a[row,:] += c[row,:] / max(||c[row,:]||2, 1e-12). One wave per row.
__global__ void k_l2acc(const float* __restrict__ c, float* __restrict__ a,
                        int nRows) {
  const int lane = threadIdx.x & 31;
  const int wave = blockIdx.x * (blockDim.x >> 5) + (threadIdx.x >> 5);
  if (wave >= nRows) return;
  const float4 cv = ((const float4*)(c + (size_t)wave * EMBD))[lane];
  float ss = cv.x * cv.x + cv.y * cv.y + cv.z * cv.z + cv.w * cv.w;
  ss = wred_sum(ss);
  const float inv = 1.0f / fmaxf(sqrtf(ss), 1e-12f);
  float4* ap = (float4*)(a + (size_t)wave * EMBD) + lane;
  float4 av = *ap;
  av.x += cv.x * inv; av.y += cv.y * inv;
  av.z += cv.z * inv; av.w += cv.w * inv;
  *ap = av;
}

__global__ void k_selfgate(const float* __restrict__ ue, const float* __restrict__ g,
                           float* __restrict__ out, int n) {
  const int i = blockIdx.x * blockDim.x + threadIdx.x;
  if (i < n) out[i] = ue[i] * (1.0f / (1.0f + expf(-g[i])));
}

// Channel attention: softmax over the two per-row scores, blend a2/a3.
__global__ void k_chatt(const float* __restrict__ t0, const float* __restrict__ t1,
                        const float* __restrict__ a2, const float* __restrict__ a3,
                        const float* __restrict__ att, float* __restrict__ HG,
                        int nRows) {
  const int lane = threadIdx.x & 31;
  const int wave = blockIdx.x * (blockDim.x >> 5) + (threadIdx.x >> 5);
  if (wave >= nRows) return;
  const float4 at = ((const float4*)att)[lane];
  const float4 v0 = ((const float4*)(t0 + (size_t)wave * EMBD))[lane];
  const float4 v1 = ((const float4*)(t1 + (size_t)wave * EMBD))[lane];
  const float w0 = wred_sum(at.x * v0.x + at.y * v0.y + at.z * v0.z + at.w * v0.w);
  const float w1 = wred_sum(at.x * v1.x + at.y * v1.y + at.z * v1.z + at.w * v1.w);
  const float m = fmaxf(w0, w1);
  const float e0 = expf(w0 - m), e1 = expf(w1 - m);
  const float s0 = e0 / (e0 + e1), s1 = 1.0f - s0;
  const float4 x0 = ((const float4*)(a2 + (size_t)wave * EMBD))[lane];
  const float4 x1 = ((const float4*)(a3 + (size_t)wave * EMBD))[lane];
  float4 o;
  o.x = s0 * x0.x + s1 * x1.x; o.y = s0 * x0.y + s1 * x1.y;
  o.z = s0 * x0.z + s1 * x1.z; o.w = s0 * x0.w + s1 * x1.w;
  ((float4*)(HG + (size_t)wave * EMBD))[lane] = o;
}

// out[row,:] = LayerNorm(table[seq[row],:]); one wave per row, 128 cols.
__global__ void k_gatherln(const int* __restrict__ seq, const float* __restrict__ tab,
                           const float* __restrict__ gamma, const float* __restrict__ beta,
                           float* __restrict__ out, int nRows) {
  const int lane = threadIdx.x & 31;
  const int wave = blockIdx.x * (blockDim.x >> 5) + (threadIdx.x >> 5);
  if (wave >= nRows) return;
  const int idx = seq[wave];
  const float4 x = ((const float4*)(tab + (size_t)idx * EMBD))[lane];
  const float mean = wred_sum(x.x + x.y + x.z + x.w) * (1.0f / 128.0f);
  const float dx = x.x - mean, dy = x.y - mean, dz = x.z - mean, dw = x.w - mean;
  const float var = wred_sum(dx * dx + dy * dy + dz * dz + dw * dw) * (1.0f / 128.0f);
  const float inv = rsqrtf(var + 1e-5f);
  const float4 g4 = ((const float4*)gamma)[lane];
  const float4 b4 = ((const float4*)beta)[lane];
  float4 o;
  o.x = dx * inv * g4.x + b4.x; o.y = dy * inv * g4.y + b4.y;
  o.z = dz * inv * g4.z + b4.z; o.w = dw * inv * g4.w + b4.w;
  ((float4*)(out + (size_t)wave * EMBD))[lane] = o;
}

// GLU epilogue: y = LN256(pre); out = y[:,:128] * relu(y[:,128:]).
__global__ void k_glu(const float* __restrict__ y, const float* __restrict__ gamma,
                      const float* __restrict__ beta, float* __restrict__ out,
                      int nRows) {
  const int lane = threadIdx.x & 31;
  const int wave = blockIdx.x * (blockDim.x >> 5) + (threadIdx.x >> 5);
  if (wave >= nRows) return;
  const float4* py = (const float4*)(y + (size_t)wave * 256);
  const float4 lo = py[lane], hi = py[32 + lane];
  float s = lo.x + lo.y + lo.z + lo.w + hi.x + hi.y + hi.z + hi.w;
  const float mean = wred_sum(s) * (1.0f / 256.0f);
  const float l0 = lo.x - mean, l1 = lo.y - mean, l2 = lo.z - mean, l3 = lo.w - mean;
  const float h0 = hi.x - mean, h1 = hi.y - mean, h2 = hi.z - mean, h3 = hi.w - mean;
  const float var = wred_sum(l0 * l0 + l1 * l1 + l2 * l2 + l3 * l3 +
                             h0 * h0 + h1 * h1 + h2 * h2 + h3 * h3) * (1.0f / 256.0f);
  const float inv = rsqrtf(var + 1e-5f);
  const float4 gl = ((const float4*)gamma)[lane], gh = ((const float4*)gamma)[32 + lane];
  const float4 bl = ((const float4*)beta)[lane],  bh = ((const float4*)beta)[32 + lane];
  const float v0 = l0 * inv * gl.x + bl.x, g0 = h0 * inv * gh.x + bh.x;
  const float v1 = l1 * inv * gl.y + bl.y, g1 = h1 * inv * gh.y + bh.y;
  const float v2 = l2 * inv * gl.z + bl.z, g2 = h2 * inv * gh.z + bh.z;
  const float v3 = l3 * inv * gl.w + bl.w, g3 = h3 * inv * gh.w + bh.w;
  float4 o;
  o.x = v0 * fmaxf(g0, 0.0f); o.y = v1 * fmaxf(g1, 0.0f);
  o.z = v2 * fmaxf(g2, 0.0f); o.w = v3 * fmaxf(g3, 0.0f);
  ((float4*)(out + (size_t)wave * EMBD))[lane] = o;
}

__global__ void k_add(const float* __restrict__ a, const float* __restrict__ b,
                      float* __restrict__ o, int n) {
  const int i = blockIdx.x * blockDim.x + threadIdx.x;
  if (i < n) o[i] = a[i] + b[i];
}

__global__ void k_copy(const float* __restrict__ s, float* __restrict__ d, int n) {
  const int i = blockIdx.x * blockDim.x + threadIdx.x;
  if (i < n) d[i] = s[i];
}

__global__ void k_zero(float* __restrict__ p, int n) {
  const int i = blockIdx.x * blockDim.x + threadIdx.x;
  if (i < n) p[i] = 0.0f;
}

// Fuse: s_c = tanh-projected score per channel (tf already tanh'ed),
// softmax over the 2 channels, blend cas2/text2.
__global__ void k_fuse(const float* __restrict__ tf0, const float* __restrict__ tf1,
                       const float* __restrict__ fw2, const float* __restrict__ b2,
                       const float* __restrict__ c0, const float* __restrict__ c1,
                       float* __restrict__ out, int nRows) {
  const int lane = threadIdx.x & 31;
  const int wave = blockIdx.x * (blockDim.x >> 5) + (threadIdx.x >> 5);
  if (wave >= nRows) return;
  const float4 w4 = ((const float4*)fw2)[lane];
  const float4 u0 = ((const float4*)(tf0 + (size_t)wave * EMBD))[lane];
  const float4 u1 = ((const float4*)(tf1 + (size_t)wave * EMBD))[lane];
  const float s0 = wred_sum(u0.x * w4.x + u0.y * w4.y + u0.z * w4.z + u0.w * w4.w) + b2[0];
  const float s1 = wred_sum(u1.x * w4.x + u1.y * w4.y + u1.z * w4.z + u1.w * w4.w) + b2[0];
  const float m = fmaxf(s0, s1);
  const float e0 = expf(s0 - m), e1 = expf(s1 - m);
  const float p0 = e0 / (e0 + e1), p1 = 1.0f - p0;
  const float4 x0 = ((const float4*)(c0 + (size_t)wave * EMBD))[lane];
  const float4 x1 = ((const float4*)(c1 + (size_t)wave * EMBD))[lane];
  float4 o;
  o.x = p0 * x0.x + p1 * x1.x; o.y = p0 * x0.y + p1 * x1.y;
  o.z = p0 * x0.z + p1 * x1.z; o.w = p0 * x0.w + p1 * x1.w;
  ((float4*)(out + (size_t)wave * EMBD))[lane] = o;
}

// ---------------------------------------------------------------------------
// Host orchestration
// ---------------------------------------------------------------------------
extern "C" void kernel_launch(void* const* d_in, const int* in_sizes, int n_in,
                              void* d_out, int out_size, void* d_ws, size_t ws_size,
                              hipStream_t stream) {
  (void)in_sizes; (void)n_in; (void)out_size; (void)ws_size;
  const int*   seq  = (const int*)  d_in[0];
  const float* txt  = (const float*)d_in[1];
  const int*   hgr  = (const int*)  d_in[2];
  const int*   hgc  = (const int*)  d_in[3];
  const float* hgv  = (const float*)d_in[4];
  const int*   hlr  = (const int*)  d_in[5];
  const int*   hlc  = (const int*)  d_in[6];
  const float* hlv  = (const float*)d_in[7];
  const float* ue   = (const float*)d_in[8];
  const float* gw0  = (const float*)d_in[9];
  const float* gb0  = (const float*)d_in[10];
  const float* gw1  = (const float*)d_in[11];
  const float* gb1  = (const float*)d_in[12];
  const float* att  = (const float*)d_in[13];
  const float* attm = (const float*)d_in[14];
  const float* linw = (const float*)d_in[15];
  const float* linb = (const float*)d_in[16];
  const float* lng  = (const float*)d_in[17];
  const float* lnb  = (const float*)d_in[18];
  const float* gluw = (const float*)d_in[19];
  const float* glub = (const float*)d_in[20];
  const float* glug = (const float*)d_in[21];
  const float* glbt = (const float*)d_in[22];
  const float* wq   = (const float*)d_in[23];
  const float* wk   = (const float*)d_in[24];
  const float* wv   = (const float*)d_in[25];
  const float* wo   = (const float*)d_in[26];
  const float* bo   = (const float*)d_in[27];
  const float* fw1  = (const float*)d_in[28];
  const float* fb1  = (const float*)d_in[29];
  const float* fw2  = (const float*)d_in[30];
  const float* fb2  = (const float*)d_in[31];
  float* outp = (float*)d_out;

  const size_t NF  = (size_t)NNODE * EMBD;           // 6.40 M floats
  const size_t SF  = (size_t)BS * EMBD;              // 1.64 M floats
  const size_t SCF = (size_t)BQ * 4 * SROW * SLD;    // 11.93 M floats
  const int    NE  = 800000;

  float* ws   = (float*)d_ws;
  float* TEXT = ws;                                  // persists
  float* HG   = ws + NF;                             // gemm tmp, then final HG
  float* P[6];
  for (int i = 0; i < 6; ++i) P[i] = ws + 2 * NF + (size_t)i * NF;

  // Sequence-phase buffers alias the structure pool (stream-ordered reuse).
  float* SC     = P[0];
  float* base   = P[0] + SCF;
  float* casn   = base + 0 * SF;
  float* textn  = base + 1 * SF;
  float* Qb     = base + 2 * SF;
  float* Kb     = base + 3 * SF;
  float* Vb     = base + 4 * SF;
  float* Ob     = base + 5 * SF;
  float* ct     = base + 6 * SF;
  float* tc     = base + 7 * SF;
  float* cas2   = base + 8 * SF;
  float* text2  = base + 9 * SF;
  float* casglu = base + 10 * SF;
  float* tf0    = base + 11 * SF;
  float* tf1    = base + 12 * SF;
  float* gtmp   = base + 13 * SF;
  float* glu256 = base + 14 * SF;                    // [BS, 256] (2*SF)

  auto gemm = [&](const float* A, int lda, const float* Bm, int ldb,
                  const float* bias, float* C, int ldc, int M, int N, int K,
                  int act) {
    const int tiles = (M / 16) * (N / 16);
    k_gemm<<<(tiles + 3) / 4, 128, 0, stream>>>(A, lda, Bm, ldb, bias, C, ldc,
                                                M, N, K, act);
  };
  auto rowk4 = [&](int nRows) { return dim3((nRows + 3) / 4); };

  // ---- Text projection: dominant GEMM (50000 x 768 x 128) ----
  gemm(txt, 768, linw, EMBD, linb, TEXT, EMBD, NNODE, EMBD, 768, 0);

  // ---- Structure embed: self-gating ----
  gemm(ue, EMBD, gw0, EMBD, gb0, HG, EMBD, NNODE, EMBD, EMBD, 0);
  k_selfgate<<<((int)NF + 255) / 256, 256, 0, stream>>>(ue, HG, P[0], (int)NF);
  gemm(ue, EMBD, gw1, EMBD, gb1, HG, EMBD, NNODE, EMBD, EMBD, 0);
  k_selfgate<<<((int)NF + 255) / 256, 256, 0, stream>>>(ue, HG, P[1], (int)NF);
  k_copy<<<((int)NF + 255) / 256, 256, 0, stream>>>(P[0], P[2], (int)NF);  // a2
  k_copy<<<((int)NF + 255) / 256, 256, 0, stream>>>(P[1], P[3], (int)NF);  // a3

  // ---- 2 propagation layers: SpMM + L2-normalized accumulate ----
  float *c2 = P[0], *c2n = P[4], *c3 = P[1], *c3n = P[5];
  const int spmmThreads = NE * 32;
  for (int l = 0; l < 2; ++l) {
    k_zero<<<((int)NF + 255) / 256, 256, 0, stream>>>(c2n, (int)NF);
    k_spmm<<<(spmmThreads + 255) / 256, 256, 0, stream>>>(hgr, hgc, hgv, c2, c2n, NE);
    k_l2acc<<<rowk4(NNODE), 128, 0, stream>>>(c2n, P[2], NNODE);
    { float* t = c2; c2 = c2n; c2n = t; }
    k_zero<<<((int)NF + 255) / 256, 256, 0, stream>>>(c3n, (int)NF);
    k_spmm<<<(spmmThreads + 255) / 256, 256, 0, stream>>>(hlr, hlc, hlv, c3, c3n, NE);
    k_l2acc<<<rowk4(NNODE), 128, 0, stream>>>(c3n, P[3], NNODE);
    { float* t = c3; c3 = c3n; c3n = t; }
  }

  // ---- Channel attention -> HG ----
  gemm(P[2], EMBD, attm, EMBD, nullptr, P[4], EMBD, NNODE, EMBD, EMBD, 0);
  gemm(P[3], EMBD, attm, EMBD, nullptr, P[5], EMBD, NNODE, EMBD, EMBD, 0);
  k_chatt<<<rowk4(NNODE), 128, 0, stream>>>(P[4], P[5], P[2], P[3], att, HG, NNODE);

  // ---- Sequence phase: gather + LN ----
  k_gatherln<<<rowk4(BS), 128, 0, stream>>>(seq, HG, lng, lnb, casn, BS);
  k_gatherln<<<rowk4(BS), 128, 0, stream>>>(seq, TEXT, lng, lnb, textn, BS);

  const float iscale = 0.17677669529663687f;  // 1/sqrt(32)
  auto mha = [&](const float* q, const float* kv, float* dst, int useMask) {
    gemm(q,  EMBD, wq, EMBD, nullptr, Qb, EMBD, BS, EMBD, EMBD, 0);
    gemm(kv, EMBD, wk, EMBD, nullptr, Kb, EMBD, BS, EMBD, EMBD, 0);
    gemm(kv, EMBD, wv, EMBD, nullptr, Vb, EMBD, BS, EMBD, EMBD, 0);
    { const int w = BQ * 4 * ST * ST;
      k_scores<<<(w + 3) / 4, 128, 0, stream>>>(Qb, Kb, SC, iscale); }
    { const int w = BQ * 4 * SQ;
      k_softmax<<<(w + 3) / 4, 128, 0, stream>>>(SC, seq, useMask); }
    { const int w = BQ * 4 * ST * 2;
      k_attnout<<<(w + 3) / 4, 128, 0, stream>>>(SC, Vb, Ob); }
    gemm(Ob, EMBD, wo, EMBD, bo, dst, EMBD, BS, EMBD, EMBD, 0);
  };

  mha(casn, textn, ct, 0);                           // ct = MHA(casn, textn, textn)
  mha(textn, casn, tc, 0);                           // tc = MHA(textn, casn, casn)

  k_add<<<((int)SF + 255) / 256, 256, 0, stream>>>(casn, ct, cas2, (int)SF);

  gemm(tc, EMBD, gluw, 256, glub, glu256, 256, BS, 256, EMBD, 0);
  k_glu<<<rowk4(BS), 128, 0, stream>>>(glu256, glug, glbt, gtmp, BS);
  k_add<<<((int)SF + 255) / 256, 256, 0, stream>>>(textn, gtmp, text2, (int)SF);

  gemm(cas2, EMBD, gluw, 256, glub, glu256, 256, BS, 256, EMBD, 0);
  k_glu<<<rowk4(BS), 128, 0, stream>>>(glu256, glug, glbt, casglu, BS);

  mha(casglu, cas2, outp, 1);                        // CE -> d_out[0]

  gemm(cas2,  EMBD, fw1, EMBD, fb1, tf0, EMBD, BS, EMBD, EMBD, 1);  // tanh
  gemm(text2, EMBD, fw1, EMBD, fb1, tf1, EMBD, BS, EMBD, EMBD, 1);  // tanh
  k_fuse<<<rowk4(BS), 128, 0, stream>>>(tf0, tf1, fw2, fb2, cas2, text2,
                                        outp + SF, BS);             // d_out[1]
}
